// HungarianMatcher_29695403884993
// MI455X (gfx1250) — matrix operand order repaired and checked by reading, not compiled, not run on hardware
//
#include <hip/hip_runtime.h>
#include <hip/hip_bf16.h>

typedef __attribute__((ext_vector_type(16))) _Float16 v16h;
typedef __attribute__((ext_vector_type(8)))  float    v8f;

#define BB   4
#define QQ   100
#define NC   81
#define MM   50
#define HWS  65536
#define QT   7          // ceil(100/16)
#define MT   4          // ceil(50/16)
#define NTILE (QT*MT)   // 28
#define NKC  32         // split-K chunks
#define KCHUNK (HWS/NKC) // 2048

// fast hardware reciprocal (v_rcp_f32, ~1 ulp) — avoids IEEE div fixup chains
__device__ __forceinline__ float fastrcp(float x) { return __builtin_amdgcn_rcpf(x); }
__device__ __forceinline__ float fastsig(float x) { return fastrcp(1.f + __expf(-x)); }

// ---------------- kernel 1: per-(b,q) softplus mean + sigmoid sum ----------
__global__ __launch_bounds__(256) void rowstats_x(const float* __restrict__ X,
                                                  float* __restrict__ spMean,
                                                  float* __restrict__ sigSum) {
    int bq = blockIdx.x;                       // 0..B*Q-1
    const float* row = X + (size_t)bq * HWS;
    float sSP = 0.f, sSG = 0.f;
    for (int i = threadIdx.x; i < HWS; i += 256) {
        float x = row[i];
        float ax = fabsf(x);
        sSP += fmaxf(x, 0.f) + log1pf(__expf(-ax));   // stable softplus
        sSG += fastsig(x);                            // sigmoid
    }
    __shared__ float s1[256], s2[256];
    s1[threadIdx.x] = sSP; s2[threadIdx.x] = sSG;
    __syncthreads();
    for (int o = 128; o > 0; o >>= 1) {
        if (threadIdx.x < o) {
            s1[threadIdx.x] += s1[threadIdx.x + o];
            s2[threadIdx.x] += s2[threadIdx.x + o];
        }
        __syncthreads();
    }
    if (threadIdx.x == 0) {
        spMean[bq] = s1[0] * (1.f / (float)HWS);
        sigSum[bq] = s2[0];
    }
}

// ---------------- kernel 2: per-(b,m) target mask sum ----------------------
__global__ __launch_bounds__(256) void rowsum_t(const float* __restrict__ T,
                                                float* __restrict__ tSum) {
    int bm = blockIdx.x;                       // 0..B*M-1
    const float* row = T + (size_t)bm * HWS;
    float s = 0.f;
    for (int i = threadIdx.x; i < HWS; i += 256) s += row[i];
    __shared__ float sh[256];
    sh[threadIdx.x] = s;
    __syncthreads();
    for (int o = 128; o > 0; o >>= 1) {
        if (threadIdx.x < o) sh[threadIdx.x] += sh[threadIdx.x + o];
        __syncthreads();
    }
    if (threadIdx.x == 0) tSum[bm] = sh[0];
}

// ---------------- kernel 3: per-(b,q) softmax max + denom over 81 ----------
__global__ __launch_bounds__(128) void logit_stats(const float* __restrict__ L,
                                                   float* __restrict__ lmax,
                                                   float* __restrict__ lsum) {
    int bq = blockIdx.x;
    int tid = threadIdx.x;
    float v = (tid < NC) ? L[(size_t)bq * NC + tid] : -3.4e38f;
    __shared__ float sh[128];
    sh[tid] = v;
    __syncthreads();
    for (int o = 64; o > 0; o >>= 1) {
        if (tid < o) sh[tid] = fmaxf(sh[tid], sh[tid + o]);
        __syncthreads();
    }
    float m = sh[0];
    __syncthreads();
    sh[tid] = (tid < NC) ? __expf(v - m) : 0.f;
    __syncthreads();
    for (int o = 64; o > 0; o >>= 1) {
        if (tid < o) sh[tid] += sh[tid + o];
        __syncthreads();
    }
    if (tid == 0) { lmax[bq] = m; lsum[bq] = sh[0]; }
}

// ---------------- kernel 4: split-K WMMA GEMM (xt and sigmoid(x)t) ---------
// grid = (NKC, NTILE, B), block = 32 (one wave).
// A fragment (16x32 f16): lane r=l&15 is row; half=l>>4 selects K-interleave:
//   a[0..7]  = K = 8*half + 0..7 ; a[8..15] = K = 16 + 8*half + 0..7
// B fragment (32x16 f16): lane n=l&15 is column; same K-interleave as A.
//
// Out-of-range q-rows / m-columns are clamped to row 0 (in-bounds, finite):
// in D = A*B, row m of D depends only on row m of A and column n only on
// column n of B, so the resulting garbage lands exclusively in tile entries
// the combine kernel never reads. No zeroing needed -> straight-line loop.
__global__ __launch_bounds__(32) void mask_gemm(const float* __restrict__ X,
                                                const float* __restrict__ T,
                                                float* __restrict__ xtPart,
                                                float* __restrict__ stPart) {
    int kc   = blockIdx.x;
    int tile = blockIdx.y;
    int b    = blockIdx.z;
    int qt = tile / MT, mt = tile % MT;
    int lane = threadIdx.x;
    int half = lane >> 4;
    int r    = lane & 15;

    int rowA = qt * 16 + r;  if (rowA >= QQ) rowA = 0;   // clamp, see note
    int rowB = mt * 16 + r;  if (rowB >= MM) rowB = 0;   // clamp, see note
    const float* Abase = X + (size_t)(b * QQ + rowA) * HWS;
    const float* Bbase = T + (size_t)(b * MM + rowB) * HWS;

    v8f accX = {}; v8f accS = {};
    int k0 = kc * KCHUNK;

#pragma unroll 2
    for (int k = k0; k < k0 + KCHUNK; k += 32) {
        const float* pa = Abase + k + 8 * half;
        const float* pb = Bbase + k + 8 * half;
        __builtin_prefetch(pa + 256, 0, 1);          // global_prefetch_b8
        __builtin_prefetch(pb + 256, 0, 1);

        float4 a0 = *(const float4*)(pa +  0);
        float4 a1 = *(const float4*)(pa +  4);
        float4 a2 = *(const float4*)(pa + 16);
        float4 a3 = *(const float4*)(pa + 20);

        float4 b0 = *(const float4*)(pb +  0);
        float4 b1 = *(const float4*)(pb +  4);
        float4 b2 = *(const float4*)(pb + 16);
        float4 b3 = *(const float4*)(pb + 20);

        float af[16] = {a0.x,a0.y,a0.z,a0.w, a1.x,a1.y,a1.z,a1.w,
                        a2.x,a2.y,a2.z,a2.w, a3.x,a3.y,a3.z,a3.w};
        float bf[16] = {b0.x,b0.y,b0.z,b0.w, b1.x,b1.y,b1.z,b1.w,
                        b2.x,b2.y,b2.z,b2.w, b3.x,b3.y,b3.z,b3.w};

        v16h A, SA, Bv;
#pragma unroll
        for (int i = 0; i < 16; ++i) {
            float x = af[i];
            A[i]  = (_Float16)x;
            SA[i] = (_Float16)fastsig(x);            // v_exp + v_rcp, no div fixup
            Bv[i] = (_Float16)bf[i];
        }

        accX = __builtin_amdgcn_wmma_f32_16x16x32_f16(
                   false, A,  false, Bv, (short)0, accX, false, false);
        accS = __builtin_amdgcn_wmma_f32_16x16x32_f16(
                   false, SA, false, Bv, (short)0, accS, false, false);
    }

    // Store partial 16x16 tiles in native register layout:
    // idx = (((kc*B + b)*NTILE + tile)*8 + v)*32 + lane
    size_t base = ((((size_t)kc * BB + b) * NTILE + tile) * 8) * 32 + lane;
#pragma unroll
    for (int v = 0; v < 8; ++v) {
        xtPart[base + (size_t)v * 32] = accX[v];
        stPart[base + (size_t)v * 32] = accS[v];
    }
}

// ---------------- kernel 5: reduce split-K partials + fuse all costs -------
__global__ __launch_bounds__(256) void combine(const float* __restrict__ L,
                                               const int* __restrict__ ids,
                                               const float* __restrict__ spMean,
                                               const float* __restrict__ sigSum,
                                               const float* __restrict__ tSum,
                                               const float* __restrict__ lmax,
                                               const float* __restrict__ lsum,
                                               const float* __restrict__ xtPart,
                                               const float* __restrict__ stPart,
                                               float* __restrict__ out) {
    int idx = blockIdx.x * 256 + threadIdx.x;
    if (idx >= BB * QQ * MM) return;
    int b = idx / (QQ * MM);
    int rem = idx % (QQ * MM);
    int q = rem / MM;
    int m = rem % MM;

    // locate (q,m) inside the wave-register tile layout
    int qt = q >> 4, qr = q & 15;
    int mt = m >> 4, mc = m & 15;
    int v = qr & 7, half = qr >> 3;
    int lane = half * 16 + mc;
    int tile = qt * MT + mt;

    float xt = 0.f, st = 0.f;
#pragma unroll 4
    for (int kc = 0; kc < NKC; ++kc) {
        size_t o = ((((size_t)kc * BB + b) * NTILE + tile) * 8 + v) * 32 + lane;
        xt += xtPart[o];
        st += stPart[o];
    }

    int bq = b * QQ + q;
    int id = ids[b * MM + m];
    float prob = __expf(L[(size_t)bq * NC + id] - lmax[bq]) * fastrcp(lsum[bq]);

    float cost_class = -prob;
    float cost_mask  = spMean[bq] - xt * (1.f / (float)HWS);
    float cost_dice  = 1.f - 2.f * st * fastrcp(sigSum[bq] + tSum[b * MM + m] + 1e-6f);

    out[idx] = cost_class + cost_mask + cost_dice;   // all costs weighted 1.0
}

// ---------------- launcher -------------------------------------------------
extern "C" void kernel_launch(void* const* d_in, const int* in_sizes, int n_in,
                              void* d_out, int out_size, void* d_ws, size_t ws_size,
                              hipStream_t stream) {
    const float* logits = (const float*)d_in[0];   // [B,Q,81]
    const float* X      = (const float*)d_in[1];   // [B,Q,H,W]
    const int*   ids    = (const int*)  d_in[2];   // [B,M]
    const float* T      = (const float*)d_in[3];   // [B,M,H,W]
    float* out = (float*)d_out;                    // [B,Q,M]
    float* ws  = (float*)d_ws;

    float* spMean = ws;                 // 400
    float* sigSum = ws + 400;           // 400
    float* tSum   = ws + 800;           // 200
    float* lmax   = ws + 1000;          // 400
    float* lsum   = ws + 1400;          // 400
    float* xtPart = ws + 1800;                       // NKC*B*NTILE*256
    float* stPart = xtPart + (size_t)NKC * BB * NTILE * 256;

    rowstats_x<<<BB * QQ, 256, 0, stream>>>(X, spMean, sigSum);
    rowsum_t  <<<BB * MM, 256, 0, stream>>>(T, tSum);
    logit_stats<<<BB * QQ, 128, 0, stream>>>(logits, lmax, lsum);

    dim3 g(NKC, NTILE, BB);
    mask_gemm<<<g, 32, 0, stream>>>(X, T, xtPart, stPart);

    int n = BB * QQ * MM;
    combine<<<(n + 255) / 256, 256, 0, stream>>>(logits, ids, spMean, sigSum,
                                                 tSum, lmax, lsum,
                                                 xtPart, stPart, out);
}
